// SemanticExtractor_73839077753451
// MI455X (gfx1250) — matrix operand order repaired and checked by reading, not compiled, not run on hardware
//
#include <hip/hip_runtime.h>
#include <hip/hip_bf16.h>
#include <math.h>

#define BB    8
#define TT    2048
#define DIN   512
#define HD    384
#define H4    1536
#define KEMB  8192
#define CD    8
#define BT    (BB * TT)
#define EPS   1e-5f

// LDS B-tile geometry: 128 rows (n) x 32 cols (k) bf16, TDM-padded to 80 B/row
#define BROW  80
#define BBUF  (128 * BROW)

typedef __attribute__((ext_vector_type(16))) __bf16 v16bf;
typedef __attribute__((ext_vector_type(8)))  __bf16 v8bf;
typedef __attribute__((ext_vector_type(8)))  float  v8f;
typedef __attribute__((ext_vector_type(4)))  unsigned v4u;
typedef __attribute__((ext_vector_type(8)))  int      v8i;
typedef __attribute__((ext_vector_type(4)))  int      v4i;

// ===========================================================================
// TDM: DMA one 32(k) x 128(n) bf16 weight tile, row-major [n][k], from global
// into LDS at ldsOff, padding each 64B row to 80B. D# per CDNA5 ISA ch.8.
// This toolchain exposes the 6-arg builtin (g0, g1, g2, g3, g4, cpol).
// ===========================================================================
__device__ __forceinline__ void tdm_load_tile(const __bf16* gsrc, unsigned ldsOff,
                                              int strideElems, int nRows)
{
    const unsigned long long ga = (unsigned long long)(size_t)gsrc;
    const unsigned dim0 = (unsigned)strideElems;   // tensor_dim0 (>= tile_dim0)
    const unsigned dim1 = (unsigned)nRows;         // tensor_dim1 (>= tile_dim1)
    const v4u g0 = { 1u,                                  // count=1, user D#
                     ldsOff,                              // lds_addr
                     (unsigned)ga,                        // global_addr[31:0]
                     (unsigned)((ga >> 32) & 0x1FFFFFFull) | (2u << 30) }; // +type=2
    const v8i g1 = { (int)((1u << 16)        // data_size = 2 bytes
                         | (1u << 20)        // pad_enable
                         | (3u << 22)        // pad_interval: 16 DWORDs (64B)
                         | (3u << 25)),      // pad_amount:   4 DWORDs (16B)
                     (int)((dim0 & 0xFFFFu) << 16),                 // dim0 lo16
                     (int)((dim0 >> 16) | ((dim1 & 0xFFFFu) << 16)),// dim0 hi/dim1 lo
                     (int)(((dim1 >> 16) & 0xFFFFu) | (32u << 16)), // dim1 hi, tile_dim0=32
                     (int)128,                                      // tile_dim1=128, tile_dim2=0
                     (int)(unsigned)strideElems,                    // tensor_dim0_stride lo32
                     0, 0 };
    const v4i gz4 = { 0, 0, 0, 0 };
    const v8i gz8 = { 0, 0, 0, 0, 0, 0, 0, 0 };
    __builtin_amdgcn_tensor_load_to_lds(g0, g1, gz4, gz4, gz8, 0);
}

// 8 WMMAs (one 16-row M-fragment vs 128 cols) against LDS buffer `bufbase`
#define WMMA_8(bufbase)                                                         \
    _Pragma("unroll")                                                           \
    for (int j = 0; j < 8; ++j) {                                               \
        const char* bp = (bufbase) + (j * 16 + l15) * BROW + 16 * half;         \
        const v8bf b0 = *(const v8bf*)bp;                                       \
        const v8bf b1 = *(const v8bf*)(bp + 32);                                \
        v16bf b;                                                                \
        _Pragma("unroll")                                                       \
        for (int i = 0; i < 8; ++i) { b[i] = b0[i]; b[i + 8] = b1[i]; }         \
        acc[j] = __builtin_amdgcn_wmma_f32_16x16x32_bf16(                       \
            false, a, false, b, (short)0, acc[j], false, false);                \
    }

// ===========================================================================
// Prep kernels: one-time f32 -> bf16 conversion / repack.
// ===========================================================================
__global__ void cvt_bf16(const float* __restrict__ in, __bf16* __restrict__ out, int n)
{
    const int i = blockIdx.x * 256 + threadIdx.x;
    if (i < n) out[i] = (__bf16)in[i];
}

// enc_w [h][d][tap] f32  ->  encT [tap][h][d] bf16
__global__ void repack_enc(const float* __restrict__ w, __bf16* __restrict__ out)
{
    const int i = blockIdx.x * 256 + threadIdx.x;
    if (i >= 7 * HD * DIN) return;
    const int d   = i % DIN;
    const int h   = (i / DIN) % HD;
    const int tap = i / (DIN * HD);
    out[i] = (__bf16)w[(h * DIN + d) * 7 + tap];
}

// ---------------------------------------------------------------------------
// Encoder conv: out[m,n] = sum_tap sum_d xbf[m+tap-3,d]*encT[tap][n][d] + b[n]
// grid = (BT/128, HD/128), block = 256 (8 waves).  TDM double-buffered B.
// ---------------------------------------------------------------------------
__global__ void enc_conv_wmma(const __bf16* __restrict__ xbf,
                              const __bf16* __restrict__ encT,
                              const float* __restrict__ bias,
                              float* __restrict__ out)
{
    __shared__ __align__(16) char BsRaw[2][BBUF];
    const int m0   = blockIdx.x * 128;
    const int n0   = blockIdx.y * 128;
    const int wave = threadIdx.x >> 5;
    const int lane = threadIdx.x & 31;
    const int half = lane >> 4;
    const int l15  = lane & 15;
    const int mw   = m0 + wave * 16;
    const int t    = (mw + l15) & (TT - 1);
    const unsigned lds0 = (unsigned)(size_t)&BsRaw[0][0];

    v8f acc[8];
#pragma unroll
    for (int j = 0; j < 8; ++j) acc[j] = (v8f){};

    const int nsteps = 7 * (DIN / 32);   // 112
    if (wave == 0)
        tdm_load_tile(encT + (size_t)n0 * DIN, lds0, DIN, HD);

    for (int s = 0; s < nsteps; ++s) {
        if (wave == 0) __builtin_amdgcn_s_wait_tensorcnt(0);
        __syncthreads();
        if (wave == 0 && s + 1 < nsteps) {
            const int tapN = (s + 1) >> 4;
            const int k0N  = ((s + 1) & 15) << 5;
            tdm_load_tile(encT + ((size_t)tapN * HD + n0) * DIN + k0N,
                          lds0 + (unsigned)(((s + 1) & 1) * BBUF), DIN, HD);
        }
        const int tap   = s >> 4;
        const int k0    = (s & 15) << 5;
        const int shift = tap - 3;
        const bool valid = (unsigned)(t + shift) < (unsigned)TT;
        v8bf lo = {}, hi = {};
        if (valid) {
            const __bf16* ap = xbf + (size_t)(mw + l15 + shift) * DIN + k0 + 8 * half;
            lo = *(const v8bf*)ap;
            hi = *(const v8bf*)(ap + 16);
        }
        v16bf a;
#pragma unroll
        for (int i = 0; i < 8; ++i) { a[i] = lo[i]; a[i + 8] = hi[i]; }
        const char* buf = &BsRaw[s & 1][0];
        WMMA_8(buf)
    }
#pragma unroll
    for (int j = 0; j < 8; ++j) {
        const int n  = n0 + 16 * j + l15;
        const float bi = bias[n];
#pragma unroll
        for (int r = 0; r < 8; ++r)
            out[(size_t)(mw + 8 * half + r) * HD + n] = acc[j][r] + bi;
    }
}

// ---------------------------------------------------------------------------
// GEMM1: out = gelu(A @ W^T + b), A bf16 [BT,Kd], W bf16 [Nd,Kd], out bf16.
// ---------------------------------------------------------------------------
template <int Kd>
__global__ void gemm_bias_gelu(const __bf16* __restrict__ A,
                               const __bf16* __restrict__ W,
                               const float* __restrict__ bias,
                               __bf16* __restrict__ out, int Nd)
{
    __shared__ __align__(16) char BsRaw[2][BBUF];
    const int m0   = blockIdx.x * 128;
    const int n0   = blockIdx.y * 128;
    const int wave = threadIdx.x >> 5;
    const int lane = threadIdx.x & 31;
    const int half = lane >> 4;
    const int l15  = lane & 15;
    const int mw   = m0 + wave * 16;
    const __bf16* Abase = A + (size_t)(mw + l15) * Kd;
    const __bf16* Wbase = W + (size_t)n0 * Kd;
    const unsigned lds0 = (unsigned)(size_t)&BsRaw[0][0];

    v8f acc[8];
#pragma unroll
    for (int j = 0; j < 8; ++j) acc[j] = (v8f){};

    const int nsteps = Kd / 32;
    if (wave == 0) tdm_load_tile(Wbase, lds0, Kd, Nd);

    for (int s = 0; s < nsteps; ++s) {
        if (wave == 0) __builtin_amdgcn_s_wait_tensorcnt(0);
        __syncthreads();
        if (wave == 0 && s + 1 < nsteps)
            tdm_load_tile(Wbase + (s + 1) * 32,
                          lds0 + (unsigned)(((s + 1) & 1) * BBUF), Kd, Nd);
        const __bf16* ap = Abase + s * 32 + 8 * half;
        const v8bf lo = *(const v8bf*)ap;
        const v8bf hi = *(const v8bf*)(ap + 16);
        v16bf a;
#pragma unroll
        for (int i = 0; i < 8; ++i) { a[i] = lo[i]; a[i + 8] = hi[i]; }
        const char* buf = &BsRaw[s & 1][0];
        WMMA_8(buf)
    }
#pragma unroll
    for (int j = 0; j < 8; ++j) {
        const int n  = n0 + 16 * j + l15;
        const float bi = bias[n];
#pragma unroll
        for (int r = 0; r < 8; ++r) {
            const float xv = acc[j][r] + bi;
            const float gl = 0.5f * xv * (1.0f + erff(xv * 0.7071067811865475f));
            out[(size_t)(mw + 8 * half + r) * Nd + n] = (__bf16)gl;
        }
    }
}

// ---------------------------------------------------------------------------
// GEMM2: out = A @ W^T + b + res, A bf16 [BT,Kd], W bf16 [Nd,Kd], out f32.
// ---------------------------------------------------------------------------
template <int Kd>
__global__ void gemm_bias_res(const __bf16* __restrict__ A,
                              const __bf16* __restrict__ W,
                              const float* __restrict__ bias,
                              const float* __restrict__ res,
                              float* __restrict__ out, int Nd)
{
    __shared__ __align__(16) char BsRaw[2][BBUF];
    const int m0   = blockIdx.x * 128;
    const int n0   = blockIdx.y * 128;
    const int wave = threadIdx.x >> 5;
    const int lane = threadIdx.x & 31;
    const int half = lane >> 4;
    const int l15  = lane & 15;
    const int mw   = m0 + wave * 16;
    const __bf16* Abase = A + (size_t)(mw + l15) * Kd;
    const __bf16* Wbase = W + (size_t)n0 * Kd;
    const unsigned lds0 = (unsigned)(size_t)&BsRaw[0][0];

    v8f acc[8];
#pragma unroll
    for (int j = 0; j < 8; ++j) acc[j] = (v8f){};

    const int nsteps = Kd / 32;
    if (wave == 0) tdm_load_tile(Wbase, lds0, Kd, Nd);

    for (int s = 0; s < nsteps; ++s) {
        if (wave == 0) __builtin_amdgcn_s_wait_tensorcnt(0);
        __syncthreads();
        if (wave == 0 && s + 1 < nsteps)
            tdm_load_tile(Wbase + (s + 1) * 32,
                          lds0 + (unsigned)(((s + 1) & 1) * BBUF), Kd, Nd);
        const __bf16* ap = Abase + s * 32 + 8 * half;
        const v8bf lo = *(const v8bf*)ap;
        const v8bf hi = *(const v8bf*)(ap + 16);
        v16bf a;
#pragma unroll
        for (int i = 0; i < 8; ++i) { a[i] = lo[i]; a[i + 8] = hi[i]; }
        const char* buf = &BsRaw[s & 1][0];
        WMMA_8(buf)
    }
#pragma unroll
    for (int j = 0; j < 8; ++j) {
        const int n  = n0 + 16 * j + l15;
        const float bi = bias[n];
#pragma unroll
        for (int r = 0; r < 8; ++r) {
            const size_t o = (size_t)(mw + 8 * half + r) * Nd + n;
            out[o] = acc[j][r] + bi + res[o];
        }
    }
}

// ---------------------------------------------------------------------------
// LayerNorm over H=384, one wave per row. f32 in -> f32 out.
// ---------------------------------------------------------------------------
__global__ void ln_rows(const float* __restrict__ in, const float* __restrict__ g,
                        const float* __restrict__ b, float* __restrict__ out)
{
    const int row  = (int)((blockIdx.x * blockDim.x + threadIdx.x) >> 5);
    const int lane = threadIdx.x & 31;
    if (row >= BT) return;
    const float* p = in + (size_t)row * HD;
    float v[12], s = 0.0f, s2 = 0.0f;
#pragma unroll
    for (int i = 0; i < 12; ++i) {
        v[i] = p[lane + 32 * i];
        s += v[i]; s2 += v[i] * v[i];
    }
#pragma unroll
    for (int o = 16; o; o >>= 1) {
        s  += __shfl_xor(s, o, 32);
        s2 += __shfl_xor(s2, o, 32);
    }
    const float mean = s * (1.0f / HD);
    const float var  = s2 * (1.0f / HD) - mean * mean;
    const float rstd = rsqrtf(var + EPS);
    float* q = out + (size_t)row * HD;
#pragma unroll
    for (int i = 0; i < 12; ++i) {
        const int hh = lane + 32 * i;
        q[hh] = (v[i] - mean) * rstd * g[hh] + b[hh];
    }
}

// ---------------------------------------------------------------------------
// Fused depthwise conv (k=7, pad 3 along t) + LayerNorm -> bf16 GEMM input.
// ---------------------------------------------------------------------------
__global__ void dwln(const float* __restrict__ h, const float* __restrict__ dww,
                     const float* __restrict__ dwb, const float* __restrict__ g,
                     const float* __restrict__ bta, __bf16* __restrict__ out)
{
    __shared__ float red[2][4];
    __shared__ float mv[2];
    const int bt = blockIdx.x;
    const int t  = bt & (TT - 1);
    float val[3], s = 0.0f, s2 = 0.0f;
#pragma unroll
    for (int j = 0; j < 3; ++j) {
        const int hh = threadIdx.x + 128 * j;
        float acc = dwb[hh];
#pragma unroll
        for (int k = 0; k < 7; ++k) {
            const int tt = t + k - 3;
            if (tt >= 0 && tt < TT)
                acc += h[(size_t)(bt + k - 3) * HD + hh] * dww[hh * 7 + k];
        }
        val[j] = acc; s += acc; s2 += acc * acc;
    }
#pragma unroll
    for (int o = 16; o; o >>= 1) {
        s  += __shfl_xor(s, o, 32);
        s2 += __shfl_xor(s2, o, 32);
    }
    const int wave = threadIdx.x >> 5, lane = threadIdx.x & 31;
    if (lane == 0) { red[0][wave] = s; red[1][wave] = s2; }
    __syncthreads();
    if (threadIdx.x == 0) {
        const float ts = red[0][0] + red[0][1] + red[0][2] + red[0][3];
        const float t2 = red[1][0] + red[1][1] + red[1][2] + red[1][3];
        const float mean = ts * (1.0f / HD);
        const float var  = t2 * (1.0f / HD) - mean * mean;
        mv[0] = mean; mv[1] = rsqrtf(var + EPS);
    }
    __syncthreads();
    const float mean = mv[0], rstd = mv[1];
#pragma unroll
    for (int j = 0; j < 3; ++j) {
        const int hh = threadIdx.x + 128 * j;
        out[(size_t)bt * HD + hh] = (__bf16)((val[j] - mean) * rstd * g[hh] + bta[hh]);
    }
}

// ---------------------------------------------------------------------------
// Fused 1x1 output conv (H->8) + nearest-code argmin over 8192 embeddings.
// ---------------------------------------------------------------------------
__global__ void out_argmin(const float* __restrict__ h, const float* __restrict__ ow,
                           const float* __restrict__ ob, const float* __restrict__ emb,
                           int* __restrict__ idxOut)
{
    const int row  = (int)((blockIdx.x * blockDim.x + threadIdx.x) >> 5);
    const int lane = threadIdx.x & 31;
    if (row >= BT) return;
    float z[CD];
#pragma unroll
    for (int c = 0; c < CD; ++c) z[c] = 0.0f;
    const float* p = h + (size_t)row * HD;
#pragma unroll
    for (int i = 0; i < 12; ++i) {
        const int hh = lane + 32 * i;
        const float v = p[hh];
#pragma unroll
        for (int c = 0; c < CD; ++c) z[c] += v * ow[c * HD + hh];
    }
#pragma unroll
    for (int c = 0; c < CD; ++c) {
#pragma unroll
        for (int o = 16; o; o >>= 1) z[c] += __shfl_xor(z[c], o, 32);
        z[c] += ob[c];
    }
    float best = 3.4e38f;
    int   bidx = 0;
    for (int j = lane; j < KEMB; j += 32) {
        const float* e = emb + (size_t)j * CD;
        float d = 0.0f;
#pragma unroll
        for (int c = 0; c < CD; ++c) { const float df = z[c] - e[c]; d += df * df; }
        if (d < best) { best = d; bidx = j; }
    }
#pragma unroll
    for (int o = 16; o; o >>= 1) {
        const float obv = __shfl_xor(best, o, 32);
        const int   oiv = __shfl_xor(bidx, o, 32);
        if (obv < best || (obv == best && oiv < bidx)) { best = obv; bidx = oiv; }
    }
    if (lane == 0) idxOut[row] = bidx;
}

// ===========================================================================
extern "C" void kernel_launch(void* const* d_in, const int* in_sizes, int n_in,
                              void* d_out, int out_size, void* d_ws, size_t ws_size,
                              hipStream_t stream)
{
    const float* x        = (const float*)d_in[0];
    const float* enc_w    = (const float*)d_in[1];
    const float* enc_b    = (const float*)d_in[2];
    const float* enc_ln_g = (const float*)d_in[3];
    const float* enc_ln_b = (const float*)d_in[4];
    const float* blk_dw_w = (const float*)d_in[5];
    const float* blk_dw_b = (const float*)d_in[6];
    const float* blk_ln_g = (const float*)d_in[7];
    const float* blk_ln_b = (const float*)d_in[8];
    const float* blk_w1   = (const float*)d_in[9];
    const float* blk_b1   = (const float*)d_in[10];
    const float* blk_w2   = (const float*)d_in[11];
    const float* blk_b2   = (const float*)d_in[12];
    const float* out_w    = (const float*)d_in[13];
    const float* out_b    = (const float*)d_in[14];
    const float* emb      = (const float*)d_in[15];

    char* ws = (char*)d_ws;
    const size_t szH    = (size_t)BT * HD * sizeof(float);
    const size_t szAln  = (size_t)BT * HD * sizeof(__bf16);
    const size_t szG    = (size_t)BT * H4 * sizeof(__bf16);
    const size_t szXbf  = (size_t)BT * DIN * sizeof(__bf16);
    const size_t szEncT = (size_t)7 * HD * DIN * sizeof(__bf16);
    const size_t szW1   = (size_t)6 * H4 * HD * sizeof(__bf16);
    const size_t szW2   = (size_t)6 * HD * H4 * sizeof(__bf16);

    float*  hA   = (float*)ws;   ws += szH;
    float*  hB   = (float*)ws;   ws += szH;
    __bf16* aln  = (__bf16*)ws;  ws += szAln;
    __bf16* g    = (__bf16*)ws;  ws += szG;
    __bf16* xbf  = (__bf16*)ws;  ws += szXbf;
    __bf16* encT = (__bf16*)ws;  ws += szEncT;
    __bf16* w1bf = (__bf16*)ws;  ws += szW1;
    __bf16* w2bf = (__bf16*)ws;  ws += szW2;

    const int nX = BT * DIN, nE = 7 * HD * DIN, nW = 6 * H4 * HD;
    cvt_bf16<<<(nX + 255) / 256, 256, 0, stream>>>(x, xbf, nX);
    repack_enc<<<(nE + 255) / 256, 256, 0, stream>>>(enc_w, encT);
    cvt_bf16<<<(nW + 255) / 256, 256, 0, stream>>>(blk_w1, w1bf, nW);
    cvt_bf16<<<(nW + 255) / 256, 256, 0, stream>>>(blk_w2, w2bf, nW);

    enc_conv_wmma<<<dim3(BT / 128, HD / 128), 256, 0, stream>>>(xbf, encT, enc_b, hA);
    ln_rows<<<BT / 8, 256, 0, stream>>>(hA, enc_ln_g, enc_ln_b, hB);

    float* cur = hB;
    float* oth = hA;
    for (int i = 0; i < 6; ++i) {
        dwln<<<BT, 128, 0, stream>>>(cur,
                                     blk_dw_w + (size_t)i * HD * 7,
                                     blk_dw_b + (size_t)i * HD,
                                     blk_ln_g + (size_t)i * HD,
                                     blk_ln_b + (size_t)i * HD, aln);
        gemm_bias_gelu<HD><<<dim3(BT / 128, H4 / 128), 256, 0, stream>>>(
            aln, w1bf + (size_t)i * H4 * HD, blk_b1 + (size_t)i * H4, g, H4);
        gemm_bias_res<H4><<<dim3(BT / 128, HD / 128), 256, 0, stream>>>(
            g, w2bf + (size_t)i * HD * H4, blk_b2 + (size_t)i * HD, cur, oth, HD);
        float* t = cur; cur = oth; oth = t;
    }

    out_argmin<<<BT / 8, 256, 0, stream>>>(cur, out_w, out_b, emb, (int*)d_out);
}
